// QuadraLayer_81269371175174
// MI455X (gfx1250) — matrix operand order repaired and checked by reading, not compiled, not run on hardware
//
#include <hip/hip_runtime.h>
#include <hip/hip_bf16.h>

// Problem constants (from the reference)
#define NROWS 16384
#define CIN   512
#define COUT  1024

typedef __attribute__((ext_vector_type(16))) __bf16 v16bf;
typedef __attribute__((ext_vector_type(8)))  float  v8f;

struct Frag32B { uint4 a, b; };   // 32 bytes == one v16bf fragment

static __device__ __forceinline__ unsigned short f32_to_bf16_rne(float f) {
  unsigned int u = __float_as_uint(f);
  u += 0x7fffu + ((u >> 16) & 1u);          // round-to-nearest-even
  return (unsigned short)(u >> 16);
}
static __device__ __forceinline__ float bf16_to_f32(unsigned short h) {
  return __uint_as_float(((unsigned int)h) << 16);
}

static __device__ __forceinline__ v16bf load_frag(const unsigned short* p0,
                                                  const unsigned short* p1) {
  Frag32B t;
  t.a = *reinterpret_cast<const uint4*>(p0);  // global_load_b128
  t.b = *reinterpret_cast<const uint4*>(p1);  // global_load_b128
  return __builtin_bit_cast(v16bf, t);
}

// ---------------------------------------------------------------------------
// Pass 1: per row, split fp32 -> bf16 hi + bf16 lo (Markidis split) and
// compute sum of squares (the ||.||^2 terms). One wave32 per row.
// ---------------------------------------------------------------------------
__global__ __launch_bounds__(256) void split_rows(
    const float* __restrict__ src, unsigned short* __restrict__ hi,
    unsigned short* __restrict__ lo, float* __restrict__ sq, int rows) {
  int wave = (int)((blockIdx.x * blockDim.x + threadIdx.x) >> 5);
  int lane = threadIdx.x & 31;
  if (wave >= rows) return;
  const float* r = src + (size_t)wave * CIN;
  unsigned short* h = hi + (size_t)wave * CIN;
  unsigned short* l = lo + (size_t)wave * CIN;
  float s = 0.0f;
#pragma unroll
  for (int j = 0; j < CIN / 128; ++j) {           // 32 lanes * float4 = 128/iter
    int c = j * 128 + lane * 4;
    float4 v = *reinterpret_cast<const float4*>(r + c);
    ushort4 vh, vl;
    vh.x = f32_to_bf16_rne(v.x); vl.x = f32_to_bf16_rne(v.x - bf16_to_f32(vh.x));
    vh.y = f32_to_bf16_rne(v.y); vl.y = f32_to_bf16_rne(v.y - bf16_to_f32(vh.y));
    vh.z = f32_to_bf16_rne(v.z); vl.z = f32_to_bf16_rne(v.z - bf16_to_f32(vh.z));
    vh.w = f32_to_bf16_rne(v.w); vl.w = f32_to_bf16_rne(v.w - bf16_to_f32(vh.w));
    *reinterpret_cast<ushort4*>(h + c) = vh;
    *reinterpret_cast<ushort4*>(l + c) = vl;
    s = fmaf(v.x, v.x, s); s = fmaf(v.y, v.y, s);
    s = fmaf(v.z, v.z, s); s = fmaf(v.w, v.w, s);
  }
#pragma unroll
  for (int off = 16; off > 0; off >>= 1) s += __shfl_xor(s, off, 32);
  if (lane == 0) sq[wave] = s;
}

// ---------------------------------------------------------------------------
// Pass 2: bf16x3 WMMA GEMM + fused rsqrt epilogue.
// Block = 8 waves = 128x128 output tile; wave = 32x64 (2x4 subtiles of 16x16).
// cross = A_hi*B_hi + A_hi*B_lo + A_lo*B_hi, accumulated in f32.
// ---------------------------------------------------------------------------
__global__ __launch_bounds__(256, 1) void qd_gemm(
    const unsigned short* __restrict__ xhi, const unsigned short* __restrict__ xlo,
    const unsigned short* __restrict__ mhi, const unsigned short* __restrict__ mlo,
    const float* __restrict__ xsq, const float* __restrict__ msq,
    float* __restrict__ out) {
  const int lane = threadIdx.x & 31;
  const int wave = threadIdx.x >> 5;   // 0..7
  const int wm   = wave >> 1;          // 0..3 (M direction)
  const int wn   = wave & 1;           // 0..1 (N direction)
  const int half = lane >> 4;          // lane-half selects K sub-range
  const int l16  = lane & 15;

  const int row0 = blockIdx.x * 128 + wm * 32;   // 2 M-subtiles
  const int col0 = blockIdx.y * 128 + wn * 64;   // 4 N-subtiles

  v8f acc[2][4];
#pragma unroll
  for (int i = 0; i < 2; ++i)
#pragma unroll
    for (int j = 0; j < 4; ++j) acc[i][j] = (v8f)(0.0f);

  for (int kk = 0; kk < CIN; kk += 32) {
    // A fragments (16x32 bf16): lane<16 -> K[0..7]+K[16..23], lane>=16 -> +8
    v16bf ahi[2], alo[2];
#pragma unroll
    for (int sm = 0; sm < 2; ++sm) {
      size_t base = (size_t)(row0 + sm * 16 + l16) * CIN + kk + half * 8;
      ahi[sm] = load_frag(xhi + base, xhi + base + 16);
      alo[sm] = load_frag(xlo + base, xlo + base + 16);
    }
    // B fragments (32x16 bf16): lane<16 -> K[0..7]+K[16..23], lane>=16 -> +8;
    // B[k][n] = mean[n][k], so each lane reads a contiguous run of mean row n.
    v16bf bhi[4], blo[4];
#pragma unroll
    for (int sn = 0; sn < 4; ++sn) {
      size_t base = (size_t)(col0 + sn * 16 + l16) * CIN + kk + half * 8;
      bhi[sn] = load_frag(mhi + base, mhi + base + 16);
      blo[sn] = load_frag(mlo + base, mlo + base + 16);
    }
#pragma unroll
    for (int sm = 0; sm < 2; ++sm)
#pragma unroll
      for (int sn = 0; sn < 4; ++sn) {
        acc[sm][sn] = __builtin_amdgcn_wmma_f32_16x16x32_bf16(
            false, ahi[sm], false, bhi[sn], (short)0, acc[sm][sn], false, false);
        acc[sm][sn] = __builtin_amdgcn_wmma_f32_16x16x32_bf16(
            false, ahi[sm], false, blo[sn], (short)0, acc[sm][sn], false, false);
        acc[sm][sn] = __builtin_amdgcn_wmma_f32_16x16x32_bf16(
            false, alo[sm], false, bhi[sn], (short)0, acc[sm][sn], false, false);
      }
  }

  // Epilogue: C/D VGPR r holds (M = half*8 + r, N = l16) of each 16x16 tile.
#pragma unroll
  for (int sm = 0; sm < 2; ++sm) {
    int rbase = row0 + sm * 16 + half * 8;
    float4 xs0 = *reinterpret_cast<const float4*>(xsq + rbase);
    float4 xs1 = *reinterpret_cast<const float4*>(xsq + rbase + 4);
    float xs[8] = {xs0.x, xs0.y, xs0.z, xs0.w, xs1.x, xs1.y, xs1.z, xs1.w};
#pragma unroll
    for (int sn = 0; sn < 4; ++sn) {
      int col = col0 + sn * 16 + l16;
      float ms = msq[col];
#pragma unroll
      for (int r = 0; r < 8; ++r) {
        float d2 = fmaf(-2.0f, acc[sm][sn][r], xs[r] + ms);
        out[(size_t)(rbase + r) * COUT + col] = rsqrtf(d2);
      }
    }
  }
}

extern "C" void kernel_launch(void* const* d_in, const int* in_sizes, int n_in,
                              void* d_out, int out_size, void* d_ws, size_t ws_size,
                              hipStream_t stream) {
  const float* x    = (const float*)d_in[0];
  const float* mean = (const float*)d_in[1];
  float* out = (float*)d_out;

  // Workspace carve (≈34.1 MB): bf16 hi/lo splits + row sums of squares.
  char* ws = (char*)d_ws;
  size_t off = 0;
  unsigned short* xhi = (unsigned short*)(ws + off); off += (size_t)NROWS * CIN * 2;
  unsigned short* xlo = (unsigned short*)(ws + off); off += (size_t)NROWS * CIN * 2;
  unsigned short* mhi = (unsigned short*)(ws + off); off += (size_t)COUT * CIN * 2;
  unsigned short* mlo = (unsigned short*)(ws + off); off += (size_t)COUT * CIN * 2;
  float* xsq = (float*)(ws + off); off += (size_t)NROWS * sizeof(float);
  float* msq = (float*)(ws + off); off += (size_t)COUT * sizeof(float);
  (void)ws_size; (void)in_sizes; (void)n_in; (void)out_size;

  // Pass 1: split + norms (one wave per row; 8 waves per block).
  split_rows<<<NROWS / 8, 256, 0, stream>>>(x, xhi, xlo, xsq, NROWS);
  split_rows<<<COUT / 8, 256, 0, stream>>>(mean, mhi, mlo, msq, COUT);

  // Pass 2: WMMA GEMM + fused rsqrt epilogue. Grid is exactly divisible.
  dim3 grid(NROWS / 128, COUT / 128);
  qd_gemm<<<grid, 256, 0, stream>>>(xhi, xlo, mhi, mlo, xsq, msq, out);
}
